// ComboLoss_69131793596443
// MI455X (gfx1250) — matrix operand order repaired and checked by reading, not compiled, not run on hardware
//
#include <hip/hip_runtime.h>

// Problem constants (reference: n=4096, d=512, f32 in, scalar f32 out)
#define N_ROWS 4096
#define DIM    512
#define BLKM   64     // rows of the I panels per workgroup
#define BLKN   128    // rows of the J panels per workgroup (output tile = 64x128)
#define KC     32     // K chunk per LDS stage (= WMMA K depth)
#define LSTR   40     // LDS row stride in bf16 elements (80B rows keep 16B alignment)

// element offsets of the four panels inside one LDS buffer
#define XI_OFF 0
#define YI_OFF (BLKM * LSTR)                 // 2560
#define XJ_OFF (2 * BLKM * LSTR)             // 5120
#define YJ_OFF (2 * BLKM * LSTR + BLKN * LSTR) // 10240
#define BUF_ELEMS (2 * BLKM * LSTR + 2 * BLKN * LSTR) // 15360 bf16 = 30720 B

typedef __attribute__((ext_vector_type(16))) __bf16 v16bf;
typedef __attribute__((ext_vector_type(8)))  float  v8f;

union V16U { uint4 u[2]; v16bf v; };

// A-matrix (16x32 bf16) fragment, CDNA5 layout:
// lanes 0-15: row m=lane, K 0..7 then 16..23 ; lanes 16-31: row m=lane-16, K 8..15 then 24..31
__device__ __forceinline__ v16bf ld_a(const __bf16* lds, int row, int kh) {
  const __bf16* p = lds + row * LSTR + kh * 8;
  V16U t;
  t.u[0] = *(const uint4*)(p);
  t.u[1] = *(const uint4*)(p + 16);
  return t.v;
}
// B-matrix (32x16 bf16) fragment: lanes 0-15: col n=lane, K 0..7 then 16..23 ; lanes 16-31: K 8..15 then 24..31.
// Gram product -> B columns are rows of the J panel (contiguous, no transpose).
__device__ __forceinline__ v16bf ld_b(const __bf16* lds, int row, int kh) {
  const __bf16* p = lds + row * LSTR + kh * 8;
  V16U t;
  t.u[0] = *(const uint4*)(p);
  t.u[1] = *(const uint4*)(p + 16);
  return t.v;
}

// CDNA5 async global->LDS copy (16B), tracked with ASYNCcnt (no VGPR staging).
__device__ __forceinline__ void async_cp(unsigned lds_addr, const __bf16* gaddr) {
  asm volatile("global_load_async_to_lds_b128 %0, %1, off"
               :: "v"(lds_addr), "v"(gaddr)
               : "memory");
}
__device__ __forceinline__ void wait_async0() {
  asm volatile("s_wait_asynccnt 0x0" ::: "memory");
}

__device__ __forceinline__ void stage(__bf16* buf,
                                      const __bf16* gXI, const __bf16* gYI,
                                      const __bf16* gXJ, const __bf16* gYJ,
                                      int a, int c4, int k0) {
  // each thread moves 6x16B: 1 row-slot of XI,YI and 2 row-slots of XJ,YJ
  const unsigned base = (unsigned)(uintptr_t)buf;        // LDS aperture: low 32 bits
  const unsigned lo = (unsigned)((a * LSTR + c4 * 8) * 2);
  const int ge = a * DIM + c4 * 8 + k0;
  async_cp(base + (unsigned)(XI_OFF * 2) + lo, gXI + ge);
  async_cp(base + (unsigned)(YI_OFF * 2) + lo, gYI + ge);
  async_cp(base + (unsigned)(XJ_OFF * 2) + lo, gXJ + ge);
  async_cp(base + (unsigned)(YJ_OFF * 2) + lo, gYJ + ge);
  const unsigned lo2 = lo + (unsigned)(BLKM * LSTR * 2);
  const int ge2 = ge + BLKM * DIM;
  async_cp(base + (unsigned)(XJ_OFF * 2) + lo2, gXJ + ge2);
  async_cp(base + (unsigned)(YJ_OFF * 2) + lo2, gYJ + ge2);
}

__global__ __launch_bounds__(256)
void prep_kernel(const float* __restrict__ x, const float* __restrict__ y,
                 __bf16* __restrict__ Xb, __bf16* __restrict__ Yb,
                 float* __restrict__ x2, float* __restrict__ y2,
                 float* __restrict__ acc) {
  __shared__ float red[512];
  const int row = blockIdx.x, tid = threadIdx.x;
  float sx = 0.f, sy = 0.f;
  for (int c = tid; c < DIM; c += 256) {
    const size_t idx = (size_t)row * DIM + c;
    const __bf16 bx = (__bf16)x[idx];
    const __bf16 by = (__bf16)y[idx];
    Xb[idx] = bx; Yb[idx] = by;
    const float fx = (float)bx, fy = (float)by;
    sx += fx * fx; sy += fy * fy;
  }
  red[tid] = sx; red[256 + tid] = sy;
  __syncthreads();
  for (int s = 128; s > 0; s >>= 1) {
    if (tid < s) { red[tid] += red[tid + s]; red[256 + tid] += red[256 + tid + s]; }
    __syncthreads();
  }
  if (tid == 0) {
    x2[row] = red[0]; y2[row] = red[256];
    if (row == 0) { acc[0] = 0.f; acc[1] = 0.f; }  // re-zeroed every call (graph-replay safe)
  }
}

__global__ __launch_bounds__(256)
void gram_fused(const __bf16* __restrict__ Xb, const __bf16* __restrict__ Yb,
                const float* __restrict__ x2, const float* __restrict__ y2,
                float* __restrict__ acc) {
  __shared__ __attribute__((aligned(16))) __bf16 lds[2][BUF_ELEMS]; // 2 x 30720 B
  __shared__ float red[512];

  const int bi = blockIdx.y, bj = blockIdx.x;
  const int tid = threadIdx.x;
  const int wv = tid >> 5, lane = tid & 31;
  const int rt = wv >> 1;          // 16-row tile 0..3 within the I panel
  const int cg = (wv & 1) * 4;     // first of four 16-col tiles within the J panel

  const __bf16* gXI = Xb + (size_t)bi * BLKM * DIM;
  const __bf16* gYI = Yb + (size_t)bi * BLKM * DIM;
  const __bf16* gXJ = Xb + (size_t)bj * BLKN * DIM;
  const __bf16* gYJ = Yb + (size_t)bj * BLKN * DIM;

  v8f aXX[4] = {{}, {}, {}, {}};
  v8f aXY[4] = {{}, {}, {}, {}};
  v8f aYY[4] = {{}, {}, {}, {}};

  const int a  = tid >> 2;   // staging row 0..63
  const int c4 = tid & 3;    // uint4 column slot within a 32-bf16 row chunk

  // prologue: fill buffer 0
  stage(&lds[0][0], gXI, gYI, gXJ, gYJ, a, c4, 0);
  wait_async0();
  __syncthreads();

  const int m  = lane & 15;
  const int kh = lane >> 4;
  constexpr int NIT = DIM / KC;

  for (int it = 0; it < NIT; ++it) {
    const int cur = it & 1;
    if (it + 1 < NIT)   // async-issue next chunk into the other buffer
      stage(&lds[cur ^ 1][0], gXI, gYI, gXJ, gYJ, a, c4, (it + 1) * KC);

    const __bf16* B = &lds[cur][0];
    const v16bf ax = ld_a(B + XI_OFF, rt * 16 + m, kh);
    const v16bf ay = ld_a(B + YI_OFF, rt * 16 + m, kh);
#pragma unroll
    for (int c = 0; c < 4; ++c) {
      const v16bf bxc = ld_b(B + XJ_OFF, (cg + c) * 16 + m, kh);
      const v16bf byc = ld_b(B + YJ_OFF, (cg + c) * 16 + m, kh);
      aXX[c] = __builtin_amdgcn_wmma_f32_16x16x32_bf16(false, ax, false, bxc, (short)0, aXX[c], false, false);
      aXY[c] = __builtin_amdgcn_wmma_f32_16x16x32_bf16(false, ax, false, byc, (short)0, aXY[c], false, false);
      aYY[c] = __builtin_amdgcn_wmma_f32_16x16x32_bf16(false, ay, false, byc, (short)0, aYY[c], false, false);
    }

    wait_async0();    // next buffer landed in LDS
    __syncthreads();  // + all waves done reading cur before it is restaged
  }

  // Epilogue: C/D layout -> element e: M = e + 8*(lane/16), N = lane%16
  float orth = 0.f, ksum = 0.f;
  const int m0  = (lane >> 4) * 8;
  const int nn  = lane & 15;
  const int gi0 = bi * BLKM + rt * 16 + m0;
#pragma unroll
  for (int ct = 0; ct < 4; ++ct) {
    const int gj = bj * BLKN + (cg + ct) * 16 + nn;
    const float x2j = x2[gj], y2j = y2[gj];
#pragma unroll
    for (int e = 0; e < 8; ++e) {
      const int gi = gi0 + e;
      const float x2i = x2[gi], y2i = y2[gi];
      const float gxx = aXX[ct][e], gxy = aXY[ct][e], gyy = aYY[ct][e];
      const float df = gxy - ((gi == gj) ? 1.f : 0.f);
      orth += df * df;                                            // orthogonal term
      ksum += __expf(-fmaxf(x2i + x2j - 2.f * gxx, 0.f));         // kx
      ksum += __expf(-fmaxf(y2i + y2j - 2.f * gyy, 0.f));         // ky
      ksum -= 2.f * __expf(-fmaxf(x2i + y2j - 2.f * gxy, 0.f));   // -2*kxy
    }
  }

  red[tid] = orth; red[256 + tid] = ksum;
  __syncthreads();
  for (int s = 128; s > 0; s >>= 1) {
    if (tid < s) { red[tid] += red[tid + s]; red[256 + tid] += red[256 + tid + s]; }
    __syncthreads();
  }
  if (tid == 0) {
    atomicAdd(&acc[0], red[0]);
    atomicAdd(&acc[1], red[256]);
  }
}

__global__ void finish_kernel(const float* __restrict__ acc, float* __restrict__ out) {
  if (threadIdx.x == 0 && blockIdx.x == 0) {
    const float inv_n2 = 1.0f / ((float)N_ROWS * (float)N_ROWS);
    out[0] = (acc[0] + acc[1]) * inv_n2;
  }
}

extern "C" void kernel_launch(void* const* d_in, const int* in_sizes, int n_in,
                              void* d_out, int out_size, void* d_ws, size_t ws_size,
                              hipStream_t stream) {
  const float* x = (const float*)d_in[0];
  const float* y = (const float*)d_in[1];
  float* out = (float*)d_out;

  char* ws = (char*)d_ws;
  __bf16* Xb = (__bf16*)ws;                                   // 4 MB
  __bf16* Yb = (__bf16*)(ws + (size_t)N_ROWS * DIM * 2);      // 4 MB
  float*  x2 = (float*)(ws + (size_t)N_ROWS * DIM * 4);       // 16 KB
  float*  y2 = x2 + N_ROWS;                                   // 16 KB
  float*  acc = y2 + N_ROWS;                                  // 2 floats

  prep_kernel<<<N_ROWS, 256, 0, stream>>>(x, y, Xb, Yb, x2, y2, acc);
  gram_fused<<<dim3(N_ROWS / BLKN, N_ROWS / BLKM), 256, 0, stream>>>(Xb, Yb, x2, y2, acc);
  finish_kernel<<<1, 64, 0, stream>>>(acc, out);
}
